// NeighborhoodAttn_1297080124168
// MI455X (gfx1250) — matrix operand order repaired and checked by reading, not compiled, not run on hardware
//
#include <hip/hip_runtime.h>
#include <hip/hip_bf16.h>
#include <hip/hip_fp16.h>

typedef __attribute__((ext_vector_type(16))) _Float16 v16h;
typedef __attribute__((ext_vector_type(8)))  float    v8f;

#define DIMC 256
#define NHEADS 8
#define HEAD_DIM 32
#define HGT 64
#define WID 64
#define NBATCH 4
#define NPIX (HGT*WID)            // 4096
#define RAD 3
#define HALO_W (16 + 2*RAD)       // 22
#define NKEYS (7*HALO_W)          // 154
#define NKT 10                    // 160 padded key columns / 16
#define ATT_SCALE 0.17677669529663687f   // 32^-0.5

// K index held by element e of a lane in half `half`: contiguous order K = half*16 + e
// (matches the B-fragment fills).
__device__ __forceinline__ int a_k_of(int e, int half) {
    return half * 16 + e;
}

// Load a B fragment (32 x 16 f16) from a row of f32 weights: lane holds column
// n = row index of w (out channel), K = c0 + half*16 + e, contiguous in memory.
__device__ __forceinline__ v16h load_b_frag_f32row(const float* wrow) {
    const float4* w4 = (const float4*)wrow;   // 64B-aligned rows
    v16h bm;
    #pragma unroll
    for (int i = 0; i < 4; ++i) {
        float4 f = w4[i];
        bm[4*i+0] = (_Float16)f.x; bm[4*i+1] = (_Float16)f.y;
        bm[4*i+2] = (_Float16)f.z; bm[4*i+3] = (_Float16)f.w;
    }
    return bm;
}

// ---------------------------------------------------------------------------
// Kernel 1: fused QKV projection.  One wave -> 16 pixels x 64 out channels
// (4 accumulators share one strided A fragment -> 4x less A traffic).
// out[b,o,pix] = sum_c x[b,c,pix] * w[o,c] + bias[o], stored f16 head-major.
// ---------------------------------------------------------------------------
__global__ __launch_bounds__(32) void qkv_proj_kernel(
    const float* __restrict__ x,
    const float* __restrict__ wq, const float* __restrict__ bq,
    const float* __restrict__ wk, const float* __restrict__ bk,
    const float* __restrict__ wv, const float* __restrict__ bv,
    _Float16* __restrict__ qh, _Float16* __restrict__ kh, _Float16* __restrict__ vh)
{
    const int lane = threadIdx.x & 31;
    const int lm = lane & 15, half = lane >> 4;
    const int mt = blockIdx.x;            // pixel tile (0..1023)
    const int nt = blockIdx.y;            // 64-wide out-channel group (0..3)
    const int which = blockIdx.z;         // 0=q 1=k 2=v

    const float* w    = which == 0 ? wq : which == 1 ? wk : wv;
    const float* bias = which == 0 ? bq : which == 1 ? bk : bv;
    _Float16*   dst   = which == 0 ? qh : which == 1 ? kh : vh;

    const int gp0  = mt * 16;
    const int b    = gp0 / NPIX;
    const int pix0 = gp0 - b * NPIX;
    const int og0  = nt * 64;

    const float* xcol = x + (size_t)b * DIMC * NPIX + pix0 + lm;

    v8f acc[4] = {{}, {}, {}, {}};
    for (int c0 = 0; c0 < DIMC; c0 += 32) {
        // L2 prefetch of next K-chunk of the strided A tile
        if (c0 + 32 < DIMC)
            __builtin_prefetch(xcol + (size_t)(c0 + 32) * NPIX, 0, 1);
        // A: X tile (16 pixels x 32 channels), channel stride NPIX (gather)
        v16h a;
        #pragma unroll
        for (int e = 0; e < 16; ++e)
            a[e] = (_Float16)xcol[(size_t)(c0 + a_k_of(e, half)) * NPIX];
        #pragma unroll
        for (int n = 0; n < 4; ++n) {
            v16h bm = load_b_frag_f32row(
                w + (size_t)(og0 + n * 16 + lm) * DIMC + c0 + half * 16);
            acc[n] = __builtin_amdgcn_wmma_f32_16x16x32_f16(false, a, false, bm,
                                                            (short)0, acc[n], false, false);
        }
    }

    #pragma unroll
    for (int n = 0; n < 4; ++n) {
        const int   o    = og0 + n * 16 + lm;
        const float bn   = bias[o];
        const int   head = o >> 5;
        const int   d    = o & 31;
        #pragma unroll
        for (int r = 0; r < 8; ++r) {
            int pix = pix0 + r + 8 * half;
            dst[(((size_t)b * NHEADS + head) * NPIX + pix) * HEAD_DIM + d] =
                (_Float16)(acc[n][r] + bn);
        }
    }
}

// ---------------------------------------------------------------------------
// Kernel 2: 7x7 neighborhood attention. One wave per (b, head, y, 16-wide x tile).
// Scores = Q(16x32) x K_halo(32x160) via 10 WMMAs; masked softmax;
// Out    = P(16x160) x V_halo(160x32) via 10 WMMAs (5 K-chunks x 2 N-tiles).
// ---------------------------------------------------------------------------
__global__ __launch_bounds__(32) void natten_kernel(
    const _Float16* __restrict__ qh, const _Float16* __restrict__ kh,
    const _Float16* __restrict__ vh, _Float16* __restrict__ oh)
{
    __shared__ __align__(16) _Float16 p_lds[16][NKT * 16];        // 16 x 160
    __shared__ __align__(16) _Float16 v_lds[NKT * 16][HEAD_DIM];  // 160 x 32

    const int lane = threadIdx.x & 31;
    const int lm = lane & 15, half = lane >> 4;

    int id = blockIdx.x;
    const int xt = id & 3;  id >>= 2;   // W/16 = 4
    const int y  = id & 63; id >>= 6;
    const int hd = id & 7;  id >>= 3;
    const int b  = id;
    const int x0 = xt * 16;

    const size_t base = ((size_t)b * NHEADS + hd) * NPIX * HEAD_DIM;

    // Q fragment: A 16(pixels) x 32(d)
    v16h qa;
    {
        const _Float16* qp = qh + base + (size_t)(y * WID + x0 + lm) * HEAD_DIM;
        #pragma unroll
        for (int e = 0; e < 16; ++e) qa[e] = qp[a_k_of(e, half)];
    }

    // Stage V halo into LDS with 128-bit copies (clamped coords; invalid keys
    // get probability 0 later so their data never matters).
    for (int k = lane; k < NKT * 16; k += 32) {
        int kc  = k < NKEYS ? k : 0;
        int dy  = kc / HALO_W;
        int hxo = kc - dy * HALO_W;
        int ky  = min(max(y + dy - RAD, 0), HGT - 1);
        int kx  = min(max(x0 - RAD + hxo, 0), WID - 1);
        const uint4* src = (const uint4*)(vh + base + (size_t)(ky * WID + kx) * HEAD_DIM);
        uint4* dstv = (uint4*)&v_lds[k][0];   // 64B per key row
        #pragma unroll
        for (int i = 0; i < 4; ++i) dstv[i] = src[i];
    }

    // Scores over 10 key tiles; C layout: lane holds (M = r+8*half, N = lm)
    float s[NKT][8];
    #pragma unroll
    for (int t = 0; t < NKT; ++t) {
        int kcol = t * 16 + lm;
        int kc   = kcol < NKEYS ? kcol : 0;
        int dy   = kc / HALO_W;
        int hxo  = kc - dy * HALO_W;
        int ky   = min(max(y + dy - RAD, 0), HGT - 1);
        int kx   = min(max(x0 - RAD + hxo, 0), WID - 1);
        const _Float16* kp = kh + base + (size_t)(ky * WID + kx) * HEAD_DIM + half * 16;
        v16h bk_;   // B: 32(d) x 16(keys); lane = key column, K = half*16 + e
        #pragma unroll
        for (int e = 0; e < 16; ++e) bk_[e] = kp[e];
        v8f c = {};
        c = __builtin_amdgcn_wmma_f32_16x16x32_f16(false, qa, false, bk_,
                                                   (short)0, c, false, false);
        #pragma unroll
        for (int r = 0; r < 8; ++r) s[t][r] = c[r];
    }

    // Mask invalid neighbors, then per-row softmax (row = query pixel M)
    #pragma unroll
    for (int t = 0; t < NKT; ++t) {
        int  kcol  = t * 16 + lm;
        bool colok = kcol < NKEYS;
        int  dy    = kcol / HALO_W;
        int  hxo   = kcol - dy * HALO_W;
        int  ky    = y + dy - RAD;
        int  kx    = x0 - RAD + hxo;
        bool inb   = colok && (ky >= 0) && (ky < HGT) && (kx >= 0) && (kx < WID);
        #pragma unroll
        for (int r = 0; r < 8; ++r) {
            int M   = r + 8 * half;
            int dxo = hxo - RAD - M;              // kx - query_x
            bool ok = inb && (dxo >= -RAD) && (dxo <= RAD);
            s[t][r] = ok ? s[t][r] * ATT_SCALE : -3.0e38f;
        }
    }
    #pragma unroll
    for (int r = 0; r < 8; ++r) {
        float m = -3.0e38f;
        #pragma unroll
        for (int t = 0; t < NKT; ++t) m = fmaxf(m, s[t][r]);
        #pragma unroll
        for (int off = 1; off < 16; off <<= 1) m = fmaxf(m, __shfl_xor(m, off, 32));
        float sum = 0.f;
        #pragma unroll
        for (int t = 0; t < NKT; ++t) { float p = __expf(s[t][r] - m); s[t][r] = p; sum += p; }
        #pragma unroll
        for (int off = 1; off < 16; off <<= 1) sum += __shfl_xor(sum, off, 32);
        float inv = 1.0f / sum;
        #pragma unroll
        for (int t = 0; t < NKT; ++t)
            p_lds[r + 8 * half][t * 16 + lm] = (_Float16)(s[t][r] * inv);
    }
    __syncthreads();

    // Out = P x V_halo : M=16 queries, N=32 d (two 16-wide tiles), K=160 (5 chunks)
    v8f oacc0 = {}, oacc1 = {};
    #pragma unroll
    for (int c = 0; c < 5; ++c) {
        v16h pa;   // A frag of P: lane row M = lm (pairs are contiguous b32 LDS reads)
        #pragma unroll
        for (int e = 0; e < 16; ++e) pa[e] = p_lds[lm][c * 32 + a_k_of(e, half)];
        v16h vb0, vb1;   // B frags of V: lane column d, K striped
        #pragma unroll
        for (int e = 0; e < 16; ++e) {
            int k  = c * 32 + half * 16 + e;
            vb0[e] = v_lds[k][lm];
            vb1[e] = v_lds[k][16 + lm];
        }
        oacc0 = __builtin_amdgcn_wmma_f32_16x16x32_f16(false, pa, false, vb0,
                                                       (short)0, oacc0, false, false);
        oacc1 = __builtin_amdgcn_wmma_f32_16x16x32_f16(false, pa, false, vb1,
                                                       (short)0, oacc1, false, false);
    }
    #pragma unroll
    for (int r = 0; r < 8; ++r) {
        int px = x0 + r + 8 * half;
        size_t addr = base + (size_t)(y * WID + px) * HEAD_DIM;
        oh[addr + lm]      = (_Float16)oacc0[r];
        oh[addr + 16 + lm] = (_Float16)oacc1[r];
    }
}

// ---------------------------------------------------------------------------
// Kernel 3: output projection, f16 head-major in -> f32 (B, DIM, H, W) out.
// One wave -> 16 pixels x 64 out channels (shared A fragment).
// ---------------------------------------------------------------------------
__global__ __launch_bounds__(32) void out_proj_kernel(
    const _Float16* __restrict__ ohh, const float* __restrict__ wo,
    const float* __restrict__ bo, float* __restrict__ out)
{
    const int lane = threadIdx.x & 31;
    const int lm = lane & 15, half = lane >> 4;
    const int mt = blockIdx.x;
    const int nt = blockIdx.y;            // 64-wide out-channel group (0..3)

    const int gp0  = mt * 16;
    const int b    = gp0 / NPIX;
    const int pix0 = gp0 - b * NPIX;
    const int og0  = nt * 64;

    v8f acc[4] = {{}, {}, {}, {}};
    for (int c0 = 0; c0 < DIMC; c0 += 32) {
        v16h a;
        #pragma unroll
        for (int e = 0; e < 16; ++e) {
            int c = c0 + a_k_of(e, half);
            int head = c >> 5, d = c & 31;
            a[e] = ohh[(((size_t)b * NHEADS + head) * NPIX + (pix0 + lm)) * HEAD_DIM + d];
        }
        #pragma unroll
        for (int n = 0; n < 4; ++n) {
            v16h bm = load_b_frag_f32row(
                wo + (size_t)(og0 + n * 16 + lm) * DIMC + c0 + half * 16);
            acc[n] = __builtin_amdgcn_wmma_f32_16x16x32_f16(false, a, false, bm,
                                                            (short)0, acc[n], false, false);
        }
    }

    #pragma unroll
    for (int n = 0; n < 4; ++n) {
        const int   o  = og0 + n * 16 + lm;
        const float bn = bo[o];
        #pragma unroll
        for (int r = 0; r < 8; ++r) {
            int pix = pix0 + r + 8 * half;
            out[((size_t)b * DIMC + o) * NPIX + pix] = acc[n][r] + bn;
        }
    }
}

extern "C" void kernel_launch(void* const* d_in, const int* in_sizes, int n_in,
                              void* d_out, int out_size, void* d_ws, size_t ws_size,
                              hipStream_t stream) {
    const float* x  = (const float*)d_in[0];
    const float* wq = (const float*)d_in[1];
    const float* bq = (const float*)d_in[2];
    const float* wk = (const float*)d_in[3];
    const float* bk = (const float*)d_in[4];
    const float* wv = (const float*)d_in[5];
    const float* bv = (const float*)d_in[6];
    const float* wo = (const float*)d_in[7];
    const float* bo = (const float*)d_in[8];

    _Float16* ws = (_Float16*)d_ws;
    const size_t per = (size_t)NBATCH * NHEADS * NPIX * HEAD_DIM;  // 4,194,304 halves
    _Float16* qh = ws;
    _Float16* kh = ws + per;
    _Float16* vh = ws + 2 * per;
    _Float16* oh = ws + 3 * per;

    const int mtiles = NBATCH * NPIX / 16;   // 1024
    dim3 g1(mtiles, DIMC / 64, 3);
    qkv_proj_kernel<<<g1, 32, 0, stream>>>(x, wq, bq, wk, bk, wv, bv, qh, kh, vh);

    natten_kernel<<<NBATCH * NHEADS * HGT * (WID / 16), 32, 0, stream>>>(qh, kh, vh, oh);

    dim3 g3(mtiles, DIMC / 64);
    out_proj_kernel<<<g3, 32, 0, stream>>>(oh, wo, bo, (float*)d_out);
}
